// SlicePredictionMappingModule_85409719648722
// MI455X (gfx1250) — compile-verified, matrix operand order checked by reading, not run on hardware
//
#include <hip/hip_runtime.h>
#include <stdint.h>

#define SLICE  262144   // C*H*W = 4*256*256
#define NSLICE 128      // B*D   = 2*64

typedef __attribute__((ext_vector_type(8))) int v8i;

__device__ __forceinline__ unsigned pack4(float f0, float f1, float f2, float f3) {
  return (f0 > 0.f ? 1u : 0u)        | ((f1 > 0.f ? 1u : 0u) << 8) |
         ((f2 > 0.f ? 1u : 0u) << 16) | ((f3 > 0.f ? 1u : 0u) << 24);
}

// ---- zero the counters in workspace (harness poisons ws; re-zero every call)
__global__ void __launch_bounds__(256) zero_ws(unsigned* ws) {
  ws[threadIdx.x] = 0u;   // 256 uints: cnt[128] + inter[128]
}

// ---- pass 1: per-slice positive counts + per-pair intersection counts.
// WMMA_I32_16x16x64_IU8 with B = ones does the 32-byte-per-lane reduction:
// every A byte of lane L lies in row L%16, so row sums = per-lane-pair sums.
__global__ void __launch_bounds__(256) slice_stats(const float* __restrict__ x,
                                                   unsigned* __restrict__ cnt,
                                                   unsigned* __restrict__ inter) {
  const int  n       = blockIdx.y;
  const int  tid     = threadIdx.x;
  const bool hasNext = (n < NSLICE - 1);
  const float* __restrict__ xn = x + (size_t)n * SLICE + (size_t)blockIdx.x * 32768;
  const float* __restrict__ xm = xn + SLICE;

  if (hasNext) __builtin_prefetch(xm + tid, 0, 0);   // global_prefetch_b8

  const v8i ones = {0x01010101, 0x01010101, 0x01010101, 0x01010101,
                    0x01010101, 0x01010101, 0x01010101, 0x01010101};
  v8i accC = {0, 0, 0, 0, 0, 0, 0, 0};
  v8i accI = {0, 0, 0, 0, 0, 0, 0, 0};

  for (int it = 0; it < 4; ++it) {
    const int seg = it * 8192;
    v8i A, AI;
#pragma unroll
    for (int j = 0; j < 8; ++j) {
      const int e = seg + tid + j * 1024;            // 4 coalesced groups of 256
      unsigned pa  = pack4(xn[e], xn[e + 256], xn[e + 512], xn[e + 768]);
      unsigned pab = 0u;
      if (hasNext) {                                  // uniform branch: EXEC stays full
        pab = pa & pack4(xm[e], xm[e + 256], xm[e + 512], xm[e + 768]);
      }
      A[j]  = (int)pa;
      AI[j] = (int)pab;
    }
    accC = __builtin_amdgcn_wmma_i32_16x16x64_iu8(false, A,  false, ones, accC, false, false);
    if (hasNext)
      accI = __builtin_amdgcn_wmma_i32_16x16x64_iu8(false, AI, false, ones, accI, false, false);
  }

  // rows 0..7 totals live (replicated) in lanes 0-15, rows 8..15 in lanes 16-31
  int sC = 0, sI = 0;
#pragma unroll
  for (int j = 0; j < 8; ++j) { sC += accC[j]; sI += accI[j]; }
  sC += __shfl_xor(sC, 16, 32);
  sI += __shfl_xor(sI, 16, 32);
  if ((tid & 31) == 0) {
    atomicAdd(&cnt[n], (unsigned)sC);
    if (hasNext) atomicAdd(&inter[n], (unsigned)sI);
  }
}

// ---- pass 2: tiny graph step over 128 scalars; folds adaptive_weight into w.
__global__ void __launch_bounds__(128) edge_weights(const unsigned* __restrict__ cnt,
                                                    const unsigned* __restrict__ inter,
                                                    const float* __restrict__ awp,
                                                    const float* __restrict__ thrp,
                                                    float* __restrict__ wout) {
  __shared__ float m[NSLICE];
  __shared__ float dinv[NSLICE];
  const int   i   = threadIdx.x;
  const float thr = thrp[0];
  const float aw  = awp[0];
  if (i < NSLICE - 1) {
    float dice = (2.0f * (float)inter[i] + 1e-5f) /
                 ((float)cnt[i] + (float)cnt[i + 1] + 1e-5f);
    m[i] = (dice > thr && dice < 1.0f) ? 1.0f : 0.0f;
  }
  __syncthreads();
  float deg = 0.0f;
  if (i < NSLICE - 1) deg += m[i];
  if (i > 0)          deg += m[i - 1];
  dinv[i] = (deg > 0.0f) ? (1.0f / sqrtf(deg)) : 0.0f;
  __syncthreads();
  if (i < NSLICE - 1) wout[i] = aw * (m[i] * dinv[i] * dinv[i + 1]);
}

// ---- pass 3: out[n] = x[n] + W[n-1]*relu(x[n-1]) + W[n]*relu(x[n+1]).
// Center slice staged to LDS with gfx1250 async loads, overlapped with the
// neighbor-slice global loads; each thread consumes only its own slots.
__global__ void __launch_bounds__(256) apply_fuse(const float* __restrict__ x,
                                                  const float* __restrict__ w,
                                                  float* __restrict__ out) {
  const int    n    = blockIdx.y;
  const int    tid  = threadIdx.x;
  const size_t base = (size_t)n * SLICE + (size_t)blockIdx.x * 4096;
  const float4* __restrict__ xc = (const float4*)(x + base);
  __shared__ float4 stage[1024];   // 16 KB

#pragma unroll
  for (int j = 0; j < 4; ++j) {
    const int slot = tid + j * 256;
    uint64_t ga = (uint64_t)(uintptr_t)(xc + slot);
    uint32_t la = (uint32_t)(uintptr_t)(&stage[slot]);   // low 32 bits = LDS offset
    asm volatile("global_load_async_to_lds_b128 %0, %1, off"
                 :: "v"(la), "v"(ga) : "memory");
  }

  const float Wp = (n > 0)          ? w[n - 1] : 0.0f;
  const float Wq = (n < NSLICE - 1) ? w[n]     : 0.0f;

  float4 p[4], q[4];
#pragma unroll
  for (int j = 0; j < 4; ++j) {
    const int slot = tid + j * 256;
    p[j] = (n > 0)          ? ((const float4*)(x + base - SLICE))[slot]
                            : make_float4(0.f, 0.f, 0.f, 0.f);
    q[j] = (n < NSLICE - 1) ? ((const float4*)(x + base + SLICE))[slot]
                            : make_float4(0.f, 0.f, 0.f, 0.f);
  }

  asm volatile("s_wait_asynccnt 0" ::: "memory");

  float4* __restrict__ o = (float4*)(out + base);
#pragma unroll
  for (int j = 0; j < 4; ++j) {
    const int slot = tid + j * 256;
    float4 c = stage[slot];
    float4 r;
    r.x = c.x + Wp * fmaxf(p[j].x, 0.f) + Wq * fmaxf(q[j].x, 0.f);
    r.y = c.y + Wp * fmaxf(p[j].y, 0.f) + Wq * fmaxf(q[j].y, 0.f);
    r.z = c.z + Wp * fmaxf(p[j].z, 0.f) + Wq * fmaxf(q[j].z, 0.f);
    r.w = c.w + Wp * fmaxf(p[j].w, 0.f) + Wq * fmaxf(q[j].w, 0.f);
    o[slot] = r;
  }
}

extern "C" void kernel_launch(void* const* d_in, const int* in_sizes, int n_in,
                              void* d_out, int out_size, void* d_ws, size_t ws_size,
                              hipStream_t stream) {
  const float* x   = (const float*)d_in[0];
  const float* aw  = (const float*)d_in[1];
  const float* thr = (const float*)d_in[2];
  float*    out   = (float*)d_out;
  unsigned* cnt   = (unsigned*)d_ws;        // [128]
  unsigned* inter = cnt + NSLICE;           // [128] (127 used)
  float*    wbuf  = (float*)(cnt + 2 * NSLICE);  // [127], aw folded in

  zero_ws<<<1, 256, 0, stream>>>(cnt);
  slice_stats<<<dim3(8, NSLICE), 256, 0, stream>>>(x, cnt, inter);
  edge_weights<<<1, 128, 0, stream>>>(cnt, inter, aw, thr, wbuf);
  apply_fuse<<<dim3(64, NSLICE), 256, 0, stream>>>(x, wbuf, out);

  (void)in_sizes; (void)n_in; (void)out_size; (void)ws_size;
}
